// ActionLSTM_61649960567164
// MI455X (gfx1250) — matrix operand order repaired and checked by reading, not compiled, not run on hardware
//
#include <hip/hip_runtime.h>

#define HID   128
#define TSTEP 128
#define NB    2048
#define NIN   68
#define KPAD  96
#define GATES 512

typedef __attribute__((ext_vector_type(16))) __bf16 v16bf;
typedef __attribute__((ext_vector_type(8)))  __bf16 v8bf;
typedef __attribute__((ext_vector_type(8)))  float  v8f;

__device__ __forceinline__ v8f splat8(float x) {
  v8f v;
#pragma unroll
  for (int e = 0; e < 8; ++e) v[e] = x;
  return v;
}

// A fragment (16x32 bf16): lane row = lane&15; elems are 16 contiguous K at k0+16*half
__device__ __forceinline__ v16bf loadA16(const __bf16* arow, int k0, int half) {
  union { v16bf v; v8bf h[2]; } u;
  u.h[0] = *(const v8bf*)(arow + k0 + 16 * half);
  u.h[1] = *(const v8bf*)(arow + k0 + 16 * half + 8);
  return u.v;
}

// B fragment (32x16 bf16): lane col = lane&15; elems are 16 contiguous K at k0+16*half
__device__ __forceinline__ v16bf loadB16(const __bf16* brow, int k0, int half) {
  const __bf16* p = brow + k0 + 16 * half;
  union { v16bf v; v8bf h[2]; } u;
  u.h[0] = *(const v8bf*)(p);
  u.h[1] = *(const v8bf*)(p + 8);
  return u.v;
}

__device__ __forceinline__ v8f wmma_bf16(v16bf a, v16bf b, v8f c) {
  return __builtin_amdgcn_wmma_f32_16x16x32_bf16(false, a, false, b, (short)0, c, false, false);
}

__device__ __forceinline__ float sigm_f(float x) { return 1.0f / (1.0f + __expf(-x)); }
__device__ __forceinline__ float tanh_f(float x) {
  float e = __expf(2.0f * x);
  return (e - 1.0f) / (e + 1.0f);
}

// ---------------------------------------------------------------------------
// Kernel 1: convert the four LSTM weight matrices fp32 -> bf16 (512x128 each)
// ---------------------------------------------------------------------------
__global__ void conv_w_kernel(const float* __restrict__ s0, const float* __restrict__ s1,
                              const float* __restrict__ s2, const float* __restrict__ s3,
                              __bf16* __restrict__ d0, __bf16* __restrict__ d1,
                              __bf16* __restrict__ d2, __bf16* __restrict__ d3) {
  int i = blockIdx.x * 256 + threadIdx.x;
  if (i < GATES * HID) {
    d0[i] = (__bf16)s0[i];
    d1[i] = (__bf16)s1[i];
    d2[i] = (__bf16)s2[i];
    d3[i] = (__bf16)s3[i];
  }
}

// ---------------------------------------------------------------------------
// Kernel 2: h = x @ fc_in_w^T + b  -> bf16 [B*T, 128].  WMMA, K padded 68->96.
// One WG = 256 threads = 8 waves handles 64 rows x 128 cols.
// ---------------------------------------------------------------------------
__global__ __launch_bounds__(256) void fcin_kernel(const float* __restrict__ x,
                                                   const float* __restrict__ w,
                                                   const float* __restrict__ bias,
                                                   __bf16* __restrict__ hout) {
  __shared__ __align__(32) __bf16 xt[64 * KPAD];
  __shared__ __align__(32) __bf16 wt[HID * KPAD];
  const int tid = threadIdx.x;
  const size_t row0 = (size_t)blockIdx.x * 64;

  for (int i = tid; i < 64 * KPAD; i += 256) {
    int r = i / KPAD, k = i - r * KPAD;
    xt[i] = (k < NIN) ? (__bf16)x[(row0 + r) * NIN + k] : (__bf16)0.0f;
  }
  for (int i = tid; i < HID * KPAD; i += 256) {
    int n = i / KPAD, k = i - n * KPAD;
    wt[i] = (k < NIN) ? (__bf16)w[n * NIN + k] : (__bf16)0.0f;
  }
  __syncthreads();

  const int wave = tid >> 5, lane = tid & 31;
  const int half = lane >> 4, ln = lane & 15;
  const int mTile = wave >> 1;
  const int nBase = (wave & 1) * 4;
  const __bf16* arow = xt + (mTile * 16 + ln) * KPAD;

#pragma unroll
  for (int nt = 0; nt < 4; ++nt) {
    const int nTile = nBase + nt;
    v8f acc = splat8(bias[nTile * 16 + ln]);
    const __bf16* brow = wt + (nTile * 16 + ln) * KPAD;
#pragma unroll
    for (int kb = 0; kb < 3; ++kb) {
      v16bf a = loadA16(arow, kb * 32, half);
      v16bf b = loadB16(brow, kb * 32, half);
      acc = wmma_bf16(a, b, acc);
    }
#pragma unroll
    for (int e = 0; e < 8; ++e) {
      int m = mTile * 16 + e + 8 * half;
      hout[(row0 + m) * HID + nTile * 16 + ln] = (__bf16)acc[e];
    }
  }
}

// ---------------------------------------------------------------------------
// Kernel 3: fused bidirectional LSTM recurrence + mean-pool accumulation.
// grid = (B/16, 2 directions), block = 256 threads (8 waves).
// Wave w computes gate tiles {w, w+8, w+16, w+24} (i/f/g/o for hidden cols
// [16w,16w+16)).  All 32 weight B-fragments are hoisted into VGPRs before the
// time loop; h_in A-fragments are software-prefetched one step ahead; h_t
// lives in a double-buffered 2x4KB LDS buffer -> one barrier per step.
// ---------------------------------------------------------------------------
__global__ __launch_bounds__(256) void lstm_kernel(
    const __bf16* __restrict__ hin,                                     // [B*T,128]
    const __bf16* __restrict__ wih_f, const __bf16* __restrict__ whh_f,
    const __bf16* __restrict__ wih_b, const __bf16* __restrict__ whh_b,
    const float* __restrict__ bih_f, const float* __restrict__ bhh_f,
    const float* __restrict__ bih_b, const float* __restrict__ bhh_b,
    float* __restrict__ pooled)                                         // [B,256] (sum over T)
{
  __shared__ __align__(32) __bf16 hbuf[2][16 * HID];

  const int tid = threadIdx.x;
  const int wave = tid >> 5, lane = tid & 31;
  const int half = lane >> 4, ln = lane & 15;
  const int b0 = blockIdx.x * 16;
  const int dir = blockIdx.y;

  const __bf16* Wih = dir ? wih_b : wih_f;
  const __bf16* Whh = dir ? whh_b : whh_f;
  const float*  Bih = dir ? bih_b : bih_f;
  const float*  Bhh = dir ? bhh_b : bhh_f;

  // ---- hoist all weight fragments into VGPRs (loop-invariant) ----
  v16bf wI[4][4], wH[4][4];
#pragma unroll
  for (int g = 0; g < 4; ++g) {
    const int grow = (g * 8 + wave) * 16 + ln;   // gate row of W  (= g*128 + 16*wave + ln)
#pragma unroll
    for (int kb = 0; kb < 4; ++kb) {
      wI[g][kb] = loadB16(Wih + (size_t)grow * HID, kb * 32, half);
      wH[g][kb] = loadB16(Whh + (size_t)grow * HID, kb * 32, half);
    }
  }

  for (int i = tid; i < 16 * HID; i += 256) {
    hbuf[0][i] = (__bf16)0.0f;
    hbuf[1][i] = (__bf16)0.0f;
  }

  float bias[4];
#pragma unroll
  for (int g = 0; g < 4; ++g) {
    int col = (g * 8 + wave) * 16 + ln;
    bias[g] = Bih[col] + Bhh[col];
  }

  v8f c    = {};
  v8f pool = {};

  const __bf16* hrow_base = hin + (size_t)(b0 + ln) * TSTEP * HID;

  // prefetch step-0 input fragments
  v16bf aIn[4];
  {
    const int t0 = dir ? (TSTEP - 1) : 0;
    const __bf16* arow_g = hrow_base + (size_t)t0 * HID;
#pragma unroll
    for (int kb = 0; kb < 4; ++kb) aIn[kb] = loadA16(arow_g, kb * 32, half);
  }

  __syncthreads();

  for (int s = 0; s < TSTEP; ++s) {
    // prefetch next step's input fragments (independent of the barrier)
    v16bf aInN[4];
    if (s + 1 < TSTEP) {
      const int tn = dir ? (TSTEP - 2 - s) : (s + 1);
      const __bf16* arow_g = hrow_base + (size_t)tn * HID;
#pragma unroll
      for (int kb = 0; kb < 4; ++kb) aInN[kb] = loadA16(arow_g, kb * 32, half);
    }

    const __bf16* arow_lds = &hbuf[s & 1][0] + ln * HID;

    v8f acc[4];
#pragma unroll
    for (int g = 0; g < 4; ++g) acc[g] = splat8(bias[g]);

#pragma unroll
    for (int kb = 0; kb < 4; ++kb) {
      v16bf aH = loadA16(arow_lds, kb * 32, half);
#pragma unroll
      for (int g = 0; g < 4; ++g) {
        acc[g] = wmma_bf16(aIn[kb], wI[g][kb], acc[g]);
        acc[g] = wmma_bf16(aH, wH[g][kb], acc[g]);
      }
    }

    v8f hv;
#pragma unroll
    for (int e = 0; e < 8; ++e) {
      float iv = sigm_f(acc[0][e]);
      float fv = sigm_f(acc[1][e]);
      float gv = tanh_f(acc[2][e]);
      float ov = sigm_f(acc[3][e]);
      float cv = fv * c[e] + iv * gv;
      c[e] = cv;
      float hh = ov * tanh_f(cv);
      hv[e] = hh;
      pool[e] += hh;
    }

    // write h_t into the *other* buffer (no WAR with this step's reads)
    __bf16* wbuf = &hbuf[(s + 1) & 1][0];
#pragma unroll
    for (int e = 0; e < 8; ++e) {
      int m = e + 8 * half;
      wbuf[m * HID + wave * 16 + ln] = (__bf16)hv[e];
    }

#pragma unroll
    for (int kb = 0; kb < 4; ++kb) aIn[kb] = aInN[kb];

    __syncthreads();   // h_t visible before next step's reads
  }

#pragma unroll
  for (int e = 0; e < 8; ++e) {
    int m = e + 8 * half;
    pooled[(size_t)(b0 + m) * 256 + dir * HID + wave * 16 + ln] = pool[e];
  }
}

// ---------------------------------------------------------------------------
// Kernel 4: out[b,c] = (pooled[b,:]/T) . fc_out_w[c,:] + fc_out_b[c]
// ---------------------------------------------------------------------------
__global__ void head_kernel(const float* __restrict__ pooled,
                            const float* __restrict__ w,
                            const float* __restrict__ b,
                            float* __restrict__ out) {
  int bi = blockIdx.x * blockDim.x + threadIdx.x;
  if (bi >= NB) return;
  const float* p = pooled + (size_t)bi * 256;
  float a0 = b[0], a1 = b[1], a2 = b[2];
  const float inv = 1.0f / (float)TSTEP;
#pragma unroll 4
  for (int k = 0; k < 256; ++k) {
    float pv = p[k] * inv;
    a0 += pv * w[k];
    a1 += pv * w[256 + k];
    a2 += pv * w[512 + k];
  }
  out[bi * 3 + 0] = a0;
  out[bi * 3 + 1] = a1;
  out[bi * 3 + 2] = a2;
}

// ---------------------------------------------------------------------------
extern "C" void kernel_launch(void* const* d_in, const int* in_sizes, int n_in,
                              void* d_out, int out_size, void* d_ws, size_t ws_size,
                              hipStream_t stream) {
  const float* x       = (const float*)d_in[0];
  const float* fc_in_w = (const float*)d_in[1];
  const float* fc_in_b = (const float*)d_in[2];
  const float* w_ih_f  = (const float*)d_in[3];
  const float* w_hh_f  = (const float*)d_in[4];
  const float* b_ih_f  = (const float*)d_in[5];
  const float* b_hh_f  = (const float*)d_in[6];
  const float* w_ih_b  = (const float*)d_in[7];
  const float* w_hh_b  = (const float*)d_in[8];
  const float* b_ih_b  = (const float*)d_in[9];
  const float* b_hh_b  = (const float*)d_in[10];
  const float* fc_out_w = (const float*)d_in[11];
  const float* fc_out_b = (const float*)d_in[12];
  float* out = (float*)d_out;

  // workspace layout (all 256B-aligned)
  char* ws = (char*)d_ws;
  const size_t szH = (size_t)NB * TSTEP * HID * sizeof(__bf16);   // 64 MiB
  const size_t szW = (size_t)GATES * HID * sizeof(__bf16);        // 128 KiB each
  __bf16* h_bf   = (__bf16*)(ws);
  __bf16* wihf16 = (__bf16*)(ws + szH);
  __bf16* whhf16 = (__bf16*)(ws + szH + szW);
  __bf16* wihb16 = (__bf16*)(ws + szH + 2 * szW);
  __bf16* whhb16 = (__bf16*)(ws + szH + 3 * szW);
  float*  pooled = (float*)(ws + szH + 4 * szW);                  // [2048,256] f32

  conv_w_kernel<<<(GATES * HID + 255) / 256, 256, 0, stream>>>(
      w_ih_f, w_hh_f, w_ih_b, w_hh_b, wihf16, whhf16, wihb16, whhb16);

  fcin_kernel<<<(NB * TSTEP) / 64, 256, 0, stream>>>(x, fc_in_w, fc_in_b, h_bf);

  dim3 grid(NB / 16, 2);
  lstm_kernel<<<grid, 256, 0, stream>>>(h_bf, wihf16, whhf16, wihb16, whhb16,
                                        b_ih_f, b_hh_f, b_ih_b, b_hh_b, pooled);

  head_kernel<<<(NB + 255) / 256, 256, 0, stream>>>(pooled, fc_out_w, fc_out_b, out);
}
